// Semantic_Space_Informed_Prompting_32779190403680
// MI455X (gfx1250) — matrix operand order repaired and checked by reading, not compiled, not run on hardware
//
#include <hip/hip_runtime.h>

typedef __bf16 v16bf __attribute__((ext_vector_type(16)));
typedef __bf16 v8bf  __attribute__((ext_vector_type(8)));
typedef float  v8f   __attribute__((ext_vector_type(8)));
typedef int    i4v   __attribute__((ext_vector_type(4)));

#define B_    32
#define DIM_  64
#define NP_   4
#define D_    1024
#define E_    768
#define VC_   32000
#define R_    (B_ * DIM_)      // 2048
#define KTOP  2
#define ROWS2 32               // r-rows per block in GEMM2 (2 subtiles of 16)
#define EPS_  1e-8f
#define NEGINF (-3.402823466e38f)

#if defined(__has_builtin)
#if __has_builtin(__builtin_amdgcn_global_load_async_to_lds_b128)
#define HAVE_ASYNC_LDS 1
#endif
#endif

// push candidate (v,i) into running top-2 (v0,i0) >= (v1,i1), ties -> lower index
__device__ __forceinline__ void push2(float v, int i,
                                      float& v0, int& i0, float& v1, int& i1) {
  if (v > v0 || (v == v0 && i < i0)) { v1 = v0; i1 = i0; v0 = v; i0 = i; }
  else if (v > v1 || (v == v1 && i < i1)) { v1 = v; i1 = i; }
}

// ---------------- f32 -> bf16 elementwise convert ----------------
__global__ void cvt_f32_bf16(const float* __restrict__ src,
                             __bf16* __restrict__ dst, int n) {
  int i = blockIdx.x * blockDim.x + threadIdx.x;
  int stride = gridDim.x * blockDim.x;
  for (; i < n; i += stride) dst[i] = (__bf16)src[i];
}

// ---------------- GEMM1: T = fe_w @ we_stored^T + fe_b ----------------
// A = fe_w bf16 [VC][E]  (M = v), B = we_stored bf16 [D][E] (N = d, K = e)
// one wave computes a 16x16 tile; block = 4 waves (4 adjacent d-tiles)
__global__ void gemm1_T(const __bf16* __restrict__ Abf,
                        const __bf16* __restrict__ Bbf,
                        const float* __restrict__ bias,
                        float* __restrict__ T, __bf16* __restrict__ Tb) {
  const int wave = threadIdx.x >> 5;
  const int lane = threadIdx.x & 31;
  const int h    = lane >> 4;
  const int l16  = lane & 15;
  const int vBase = blockIdx.x * 16;
  const int dBase = (blockIdx.y * 4 + wave) * 16;

  v8f c;
  {
    float4 b0 = *(const float4*)(bias + vBase + 8 * h);
    float4 b1 = *(const float4*)(bias + vBase + 8 * h + 4);
    c[0] = b0.x; c[1] = b0.y; c[2] = b0.z; c[3] = b0.w;
    c[4] = b1.x; c[5] = b1.y; c[6] = b1.z; c[7] = b1.w;
  }

  const __bf16* arow = Abf + (size_t)(vBase + l16) * E_;
  const __bf16* brow = Bbf + (size_t)(dBase + l16) * E_;

  for (int kb = 0; kb < E_; kb += 32) {
    v16bf a = *(const v16bf*)(arow + kb + 16 * h);
    v16bf b = *(const v16bf*)(brow + kb + 16 * h);
    c = __builtin_amdgcn_wmma_f32_16x16x32_bf16(false, a, false, b,
                                                (short)0, c, false, false);
  }

  const int d = dBase + l16;
#pragma unroll
  for (int j = 0; j < 8; ++j) {
    int v = vBase + 8 * h + j;
    size_t idx = (size_t)v * D_ + d;
    T[idx]  = c[j];
    Tb[idx] = (__bf16)c[j];
  }
}

// ---------------- w_norm[v] = max(2*||T[v,:]||, eps) ----------------
__global__ void wnorm_k(const float* __restrict__ T, float* __restrict__ wn) {
  int row  = blockIdx.x * (blockDim.x >> 5) + (threadIdx.x >> 5);
  int lane = threadIdx.x & 31;
  const float* p = T + (size_t)row * D_;
  float s = 0.f;
  for (int i = lane; i < D_; i += 32) { float x = p[i]; s += x * x; }
#pragma unroll
  for (int o = 16; o > 0; o >>= 1) s += __shfl_down(s, o, 32);
  if (lane == 0) wn[row] = fmaxf(2.0f * sqrtf(s), EPS_);
}

// ---- s = sum_np P (bf16), p_norm, and copy P into Z[:, 2:6, :] ----
__global__ void sprep_k(const float* __restrict__ P, __bf16* __restrict__ sb,
                        float* __restrict__ pn, float* __restrict__ outZ) {
  const int r   = blockIdx.x;
  const int tid = threadIdx.x;
  const float* pr = P + (size_t)r * NP_ * D_;
  float acc = 0.f;
  for (int k = tid; k < D_; k += blockDim.x) {
    float x0 = pr[0 * D_ + k], x1 = pr[1 * D_ + k];
    float x2 = pr[2 * D_ + k], x3 = pr[3 * D_ + k];
    sb[(size_t)r * D_ + k] = (__bf16)(x0 + x1 + x2 + x3);
    acc += x0 * x0 + x1 * x1 + x2 * x2 + x3 * x3;
    float* z = outZ + ((size_t)r * (NP_ + KTOP) + KTOP) * D_ + k;
    z[0 * D_] = x0; z[1 * D_] = x1; z[2 * D_] = x2; z[3 * D_] = x3;
  }
  __shared__ float red[8];
#pragma unroll
  for (int o = 16; o > 0; o >>= 1) acc += __shfl_down(acc, o, 32);
  int lane = tid & 31, w = tid >> 5;
  if (lane == 0) red[w] = acc;
  __syncthreads();
  if (tid == 0) {
    float t = 0.f;
    for (int i = 0; i < (int)(blockDim.x >> 5); ++i) t += red[i];
    pn[r] = fmaxf(sqrtf(t), EPS_);
  }
}

// ---- GEMM2 + cosine scaling + fused top-2 + e_k gather ----
// block = 256 threads (8 waves); each block owns 32 output rows r (two 16-wide
// subtiles staged in 64KB LDS); each wave computes a 16v x 32r strip per v-tile
// (one A load feeds two WMMAs), waves stride the 2000 v-tiles.
__global__ void gemm2_topk(const __bf16* __restrict__ Tb,
                           const float* __restrict__ Tf,
                           const __bf16* __restrict__ sb,
                           const float* __restrict__ wn,
                           const float* __restrict__ pn,
                           float* __restrict__ outZ, float* __restrict__ outV) {
  __shared__ __align__(32) __bf16 sS[ROWS2 * D_];   // 64 KB
  __shared__ float mv[8][2][2][16];
  __shared__ int   mi[8][2][2][16];
  __shared__ int   idxF[ROWS2][2];

  const int tid  = threadIdx.x;
  const int wave = tid >> 5;
  const int lane = tid & 31;
  const int h    = lane >> 4;
  const int l16  = lane & 15;
  const int rBase = blockIdx.x * ROWS2;

  // stage 32 contiguous s rows into LDS
  {
    const uint4* src = (const uint4*)(sb + (size_t)rBase * D_);
    uint4* dst = (uint4*)sS;
    const int n16 = (ROWS2 * D_ * 2) / 16;
#ifdef HAVE_ASYNC_LDS
    for (int i = tid; i < n16; i += blockDim.x) {
      __builtin_amdgcn_global_load_async_to_lds_b128(
          (__attribute__((address_space(1))) i4v*)(uintptr_t)(src + i),
          (__attribute__((address_space(3))) i4v*)(uintptr_t)(dst + i),
          0, 0);
    }
#if __has_builtin(__builtin_amdgcn_s_wait_asynccnt)
    __builtin_amdgcn_s_wait_asynccnt(0);
#else
    asm volatile("s_wait_asynccnt 0x0" ::: "memory");
#endif
#else
    for (int i = tid; i < n16; i += blockDim.x) dst[i] = src[i];
#endif
  }
  __syncthreads();

  const float invPn0 = 1.0f / pn[rBase + l16];
  const float invPn1 = 1.0f / pn[rBase + 16 + l16];

  float t0v[2] = { NEGINF, NEGINF };   // [subtile] best
  float t1v[2] = { NEGINF, NEGINF };   // [subtile] 2nd best
  int   t0i[2] = { -1, -1 };
  int   t1i[2] = { -1, -1 };

  for (int vt = wave; vt < VC_ / 16; vt += 8) {
    const int vBase = vt * 16;
    v8f c0 = {};
    v8f c1 = {};
    const __bf16* arow = Tb + (size_t)(vBase + l16) * D_;
    for (int kb = 0; kb < D_; kb += 32) {
      v16bf a = *(const v16bf*)(arow + kb + 16 * h);
      v16bf b0 = *(const v16bf*)(&sS[l16 * D_ + kb + 16 * h]);
      v16bf b1 = *(const v16bf*)(&sS[(16 + l16) * D_ + kb + 16 * h]);
      c0 = __builtin_amdgcn_wmma_f32_16x16x32_bf16(false, a, false, b0,
                                                   (short)0, c0, false, false);
      c1 = __builtin_amdgcn_wmma_f32_16x16x32_bf16(false, a, false, b1,
                                                   (short)0, c1, false, false);
    }
    float4 w0 = *(const float4*)(wn + vBase + 8 * h);
    float4 w1 = *(const float4*)(wn + vBase + 8 * h + 4);
    float wnj[8] = { w0.x, w0.y, w0.z, w0.w, w1.x, w1.y, w1.z, w1.w };
#pragma unroll
    for (int j = 0; j < 8; ++j) {
      int v = vBase + 8 * h + j;
      float inw = 1.0f / wnj[j];
      push2(c0[j] * invPn0 * inw, v, t0v[0], t0i[0], t1v[0], t1i[0]);
      push2(c1[j] * invPn1 * inw, v, t0v[1], t0i[1], t1v[1], t1i[1]);
    }
  }

  // merge the two lane-halves owning the same output row r (per subtile)
#pragma unroll
  for (int s = 0; s < 2; ++s) {
    float ov0 = __shfl_xor(t0v[s], 16, 32); int oi0 = __shfl_xor(t0i[s], 16, 32);
    float ov1 = __shfl_xor(t1v[s], 16, 32); int oi1 = __shfl_xor(t1i[s], 16, 32);
    push2(ov0, oi0, t0v[s], t0i[s], t1v[s], t1i[s]);
    push2(ov1, oi1, t0v[s], t0i[s], t1v[s], t1i[s]);
  }
  if (lane < 16) {
#pragma unroll
    for (int s = 0; s < 2; ++s) {
      mv[wave][s][0][lane] = t0v[s]; mi[wave][s][0][lane] = t0i[s];
      mv[wave][s][1][lane] = t1v[s]; mi[wave][s][1][lane] = t1i[s];
    }
  }
  __syncthreads();

  // final 8-wave merge, write top-2 values, keep indices for gather
  if (tid < ROWS2) {
    const int sub = tid >> 4;
    const int l   = tid & 15;
    float b0 = NEGINF, b1 = NEGINF; int i0 = -1, i1 = -1;
    for (int w = 0; w < 8; ++w) {
      push2(mv[w][sub][0][l], mi[w][sub][0][l], b0, i0, b1, i1);
      push2(mv[w][sub][1][l], mi[w][sub][1][l], b0, i0, b1, i1);
    }
    int r = rBase + tid;
    outV[r * KTOP + 0] = b0;
    outV[r * KTOP + 1] = b1;
    idxF[tid][0] = i0; idxF[tid][1] = i1;
  }
  __syncthreads();

  // gather e_k rows (f32 T) into Z[:, 0:2, :]
  for (int t = tid; t < ROWS2 * KTOP * D_; t += blockDim.x) {
    int row  = t >> 11;            // / (KTOP*D_)
    int slot = (t >> 10) & 1;
    int k    = t & (D_ - 1);
    int idx  = idxF[row][slot];
    outZ[(((size_t)(rBase + row)) * (NP_ + KTOP) + slot) * D_ + k] =
        Tf[(size_t)idx * D_ + k];
  }
}

extern "C" void kernel_launch(void* const* d_in, const int* in_sizes, int n_in,
                              void* d_out, int out_size, void* d_ws, size_t ws_size,
                              hipStream_t stream) {
  const float* P  = (const float*)d_in[0];   // [32,64,4,1024]
  const float* we = (const float*)d_in[1];   // [1024,768]
  const float* fw = (const float*)d_in[2];   // [32000,768]
  const float* fb = (const float*)d_in[3];   // [32000]

  float* outZ = (float*)d_out;                          // [2048,6,1024]
  float* outV = outZ + (size_t)R_ * (NP_ + KTOP) * D_;  // [2048,2]

  char* ws = (char*)d_ws;
  size_t off = 0;
  auto alloc = [&](size_t bytes) -> void* {
    void* p = ws + off;
    off = (off + bytes + 255) & ~(size_t)255;
    return p;
  };
  __bf16* fewb = (__bf16*)alloc((size_t)VC_ * E_ * sizeof(__bf16));
  __bf16* wesb = (__bf16*)alloc((size_t)D_ * E_ * sizeof(__bf16));
  float*  T    = (float*) alloc((size_t)VC_ * D_ * sizeof(float));
  __bf16* Tb   = (__bf16*)alloc((size_t)VC_ * D_ * sizeof(__bf16));
  __bf16* sb   = (__bf16*)alloc((size_t)R_ * D_ * sizeof(__bf16));
  float*  wn   = (float*) alloc((size_t)VC_ * sizeof(float));
  float*  pnm  = (float*) alloc((size_t)R_ * sizeof(float));

  cvt_f32_bf16<<<2048, 256, 0, stream>>>(fw, fewb, VC_ * E_);
  cvt_f32_bf16<<<1024, 256, 0, stream>>>(we, wesb, D_ * E_);

  dim3 g1(VC_ / 16, D_ / (16 * 4));
  gemm1_T<<<g1, 128, 0, stream>>>(fewb, wesb, fb, T, Tb);

  wnorm_k<<<VC_ / 4, 128, 0, stream>>>(T, wn);
  sprep_k<<<R_, 256, 0, stream>>>(P, sb, pnm, outZ);
  gemm2_topk<<<R_ / ROWS2, 256, 0, stream>>>(Tb, T, sb, wn, pnm, outZ, outV);
}
